// Mask_Rcnn_2851858285273
// MI455X (gfx1250) — matrix operand order repaired and checked by reading, not compiled, not run on hardware
//
#include <hip/hip_runtime.h>
#include <hip/hip_bf16.h>
#include <math.h>

typedef _Float16 half8  __attribute__((ext_vector_type(8)));
typedef _Float16 half16 __attribute__((ext_vector_type(16)));
typedef float    f32x8  __attribute__((ext_vector_type(8)));

#define NUM_CLS   81
#define N_ROI     512
#define N_POS     128
#define POOL      7
#define FEATC     256
#define K1        (FEATC * POOL * POOL)   /* 12544 */
#define KT1       (K1 / 32)               /* 392 (even) */
#define HID       1024
#define KT2       (HID / 32)              /* 32  (even) */
#define NCLS_PAD  128                     /* 8 tiles  -> 2 groups of 4 */
#define NLOC_PAD  384                     /* 24 tiles -> 6 groups of 4 */

// A fragment: halves 0..15 contiguous at p (k = 16*(lane>>4) + i)
static __device__ __forceinline__ half16 mrcnn_lda(const _Float16* p) {
    half8 lo = *(const half8*)(p);
    half8 hi = *(const half8*)(p + 8);
    return __builtin_shufflevector(lo, hi, 0,1,2,3,4,5,6,7,8,9,10,11,12,13,14,15);
}
// B fragment: 16 contiguous halves per lane (fragment-packed layout)
static __device__ __forceinline__ half16 mrcnn_ldb(const _Float16* p) {
    half8 lo = *(const half8*)(p);
    half8 hi = *(const half8*)(p + 8);
    return __builtin_shufflevector(lo, hi, 0,1,2,3,4,5,6,7,8,9,10,11,12,13,14,15);
}

// ---------------------------------------------------------------------------
// ROI level select + ROI-Align, writing x directly as f16 row-major (N, C*49)
// one block per ROI, thread = channel
// ---------------------------------------------------------------------------
__global__ void mrcnn_roi_align_kernel(const float* __restrict__ P2,
                                       const float* __restrict__ P3,
                                       const float* __restrict__ P4,
                                       const float* __restrict__ P5,
                                       const float* __restrict__ rois,
                                       _Float16* __restrict__ x16)
{
    const int n = blockIdx.x;
    const int c = threadIdx.x;

    float rx1 = rois[n * 4 + 0], ry1 = rois[n * 4 + 1];
    float rx2 = rois[n * 4 + 2], ry2 = rois[n * 4 + 3];

    float area = (rx2 - rx1 + 1.f) * (ry2 - ry1 + 1.f);
    float lf = floorf(4.f + log2f(sqrtf(area) / 224.f));
    lf = fminf(fmaxf(lf, 2.f), 5.f);
    int lv = (int)lf - 2;

    const float* feat; int H, W; float scale;
    switch (lv) {
        case 0:  feat = P2; H = 200; W = 256; scale = 0.25f;    break;
        case 1:  feat = P3; H = 100; W = 128; scale = 0.125f;   break;
        case 2:  feat = P4; H = 50;  W = 64;  scale = 0.0625f;  break;
        default: feat = P5; H = 25;  W = 32;  scale = 0.03125f; break;
    }

    float x1 = rx1 * scale, y1 = ry1 * scale;
    float x2 = rx2 * scale, y2 = ry2 * scale;
    float bw = fmaxf(x2 - x1, 1.f) * (1.f / POOL);
    float bh = fmaxf(y2 - y1, 1.f) * (1.f / POOL);
    float Wf = (float)W, Hf = (float)H;

    const float* fc = feat + (size_t)c * H * W;

    for (int i = 0; i < POOL; ++i) {
        for (int j = 0; j < POOL; ++j) {
            float acc = 0.f;
            for (int sy = 0; sy < 2; ++sy) {
                float py = y1 + ((float)i + (sy + 0.5f) * 0.5f) * bh;
                for (int sx = 0; sx < 2; ++sx) {
                    float px = x1 + ((float)j + (sx + 0.5f) * 0.5f) * bw;
                    bool valid = (px > -1.f) && (px < Wf) && (py > -1.f) && (py < Hf);
                    float xx = fminf(fmaxf(px, 0.f), Wf - 1.f);
                    float yy = fminf(fmaxf(py, 0.f), Hf - 1.f);
                    float x0f = floorf(xx), y0f = floorf(yy);
                    float lx = xx - x0f, ly = yy - y0f;
                    int x0 = (int)x0f, y0 = (int)y0f;
                    int x1i = min(x0 + 1, W - 1);
                    int y1i = min(y0 + 1, H - 1);
                    float v00 = fc[y0  * W + x0 ];
                    float v01 = fc[y0  * W + x1i];
                    float v10 = fc[y1i * W + x0 ];
                    float v11 = fc[y1i * W + x1i];
                    float v = v00 * (1.f - ly) * (1.f - lx)
                            + v01 * (1.f - ly) * lx
                            + v10 * ly * (1.f - lx)
                            + v11 * ly * lx;
                    acc += valid ? v : 0.f;
                }
            }
            x16[((size_t)n * FEATC + c) * 49 + i * 7 + j] = (_Float16)(acc * 0.25f);
        }
    }
}

// ---------------------------------------------------------------------------
// Pack f32 weights (K x n_real, row-major) into WMMA B-fragment-ordered f16:
//   out[((nt*Kt + kt)*32 + lane)*16 + i]  where
//   n = nt*16 + (lane&15),  k = kt*32 + 16*(lane>>4) + i
// Zero-pads columns n >= n_real.
// ---------------------------------------------------------------------------
__global__ void mrcnn_pack_w_kernel(const float* __restrict__ Wsrc,
                                    _Float16* __restrict__ out,
                                    int Kt, int Nt, int n_real)
{
    size_t total = (size_t)Kt * Nt * 512;
    size_t stride = (size_t)gridDim.x * blockDim.x;
    for (size_t p = (size_t)blockIdx.x * blockDim.x + threadIdx.x; p < total; p += stride) {
        int    i    = (int)(p & 15);
        int    lane = (int)((p >> 4) & 31);
        size_t tile = p >> 9;
        int    kt   = (int)(tile % (size_t)Kt);
        int    nt   = (int)(tile / (size_t)Kt);
        int n = nt * 16 + (lane & 15);
        int k = kt * 32 + ((lane >> 4) << 4) + i;
        float v = (n < n_real) ? Wsrc[(size_t)k * n_real + n] : 0.f;
        out[p] = (_Float16)v;
    }
}

// ---------------------------------------------------------------------------
// Register-blocked WMMA GEMM: wave computes a 16(M) x 64(N) tile via
// 4 f32 accumulators sharing one A fragment per K-step.
// K-loop is unrolled x2 with ping-pong fragment sets (X/Y) so loads for the
// next step are in flight in DIFFERENT registers while WMMAs execute.
// Requires Kt even (all call sites: 392, 32).  Tail over-reads <= 2 KB of
// workspace (never fed to WMMA) -- every A/B region is followed by more ws.
// ---------------------------------------------------------------------------
__global__ void mrcnn_wmma_gemm_kernel(const _Float16* __restrict__ A,
                                       const _Float16* __restrict__ Bp,
                                       const float* __restrict__ bias,
                                       float* __restrict__ out32,
                                       _Float16* __restrict__ out16,
                                       int Mt, int Ntg, int Kt, int K,
                                       int N_out, int n_real, int relu)
{
    int wave  = blockIdx.x * (blockDim.x >> 5) + (threadIdx.x >> 5);
    int total = Mt * Ntg;
    if (wave >= total) return;

    int ng = wave % Ntg;          // group of 4 consecutive n-tiles
    int mt = wave / Ntg;          // consecutive waves share A rows
    int nt0 = ng * 4;
    int lane = threadIdx.x & 31;
    int s = lane >> 4;

    int mrow = mt * 16 + (lane & 15);
    const _Float16* Ap = A + (size_t)mrow * K + 16 * s;    // 16B aligned

    size_t gstride = (size_t)Kt * 512;                     // halves per n-tile column
    const _Float16* B0 = Bp + (size_t)nt0 * gstride + (size_t)lane * 16;
    const _Float16* B1 = B0 + gstride;
    const _Float16* B2 = B1 + gstride;
    const _Float16* B3 = B2 + gstride;

    f32x8 acc0 = {}, acc1 = {}, acc2 = {}, acc3 = {};

    // prologue: fragment set X for kt=0
    half16 aX  = mrcnn_lda(Ap);
    half16 bX0 = mrcnn_ldb(B0);
    half16 bX1 = mrcnn_ldb(B1);
    half16 bX2 = mrcnn_ldb(B2);
    half16 bX3 = mrcnn_ldb(B3);

    for (int kt = 0; kt < Kt; kt += 2) {
        // loads for kt+1 (set Y) -- in flight during X WMMAs
        half16 aY  = mrcnn_lda(Ap + 32);
        half16 bY0 = mrcnn_ldb(B0 + 512);
        half16 bY1 = mrcnn_ldb(B1 + 512);
        half16 bY2 = mrcnn_ldb(B2 + 512);
        half16 bY3 = mrcnn_ldb(B3 + 512);
        __builtin_prefetch(B0 + 1024, 0, 1);
        __builtin_prefetch(B1 + 1024, 0, 1);
        __builtin_prefetch(B2 + 1024, 0, 1);
        __builtin_prefetch(B3 + 1024, 0, 1);

        acc0 = __builtin_amdgcn_wmma_f32_16x16x32_f16(false, aX, false, bX0, (short)0, acc0, false, false);
        acc1 = __builtin_amdgcn_wmma_f32_16x16x32_f16(false, aX, false, bX1, (short)0, acc1, false, false);
        acc2 = __builtin_amdgcn_wmma_f32_16x16x32_f16(false, aX, false, bX2, (short)0, acc2, false, false);
        acc3 = __builtin_amdgcn_wmma_f32_16x16x32_f16(false, aX, false, bX3, (short)0, acc3, false, false);

        // loads for kt+2 (set X) -- in flight during Y WMMAs
        // (last iteration loads harmless garbage that is never consumed)
        aX  = mrcnn_lda(Ap + 64);
        bX0 = mrcnn_ldb(B0 + 1024);
        bX1 = mrcnn_ldb(B1 + 1024);
        bX2 = mrcnn_ldb(B2 + 1024);
        bX3 = mrcnn_ldb(B3 + 1024);

        acc0 = __builtin_amdgcn_wmma_f32_16x16x32_f16(false, aY, false, bY0, (short)0, acc0, false, false);
        acc1 = __builtin_amdgcn_wmma_f32_16x16x32_f16(false, aY, false, bY1, (short)0, acc1, false, false);
        acc2 = __builtin_amdgcn_wmma_f32_16x16x32_f16(false, aY, false, bY2, (short)0, acc2, false, false);
        acc3 = __builtin_amdgcn_wmma_f32_16x16x32_f16(false, aY, false, bY3, (short)0, acc3, false, false);

        Ap += 64;
        B0 += 1024; B1 += 1024; B2 += 1024; B3 += 1024;
    }

    #pragma unroll
    for (int g = 0; g < 4; ++g) {
        f32x8 c = (g == 0) ? acc0 : (g == 1) ? acc1 : (g == 2) ? acc2 : acc3;
        int n = (nt0 + g) * 16 + (lane & 15);
        float bv = (n < n_real) ? bias[n] : 0.f;
        #pragma unroll
        for (int r = 0; r < 8; ++r) {
            int m = mt * 16 + r + 8 * s;
            float v = c[r] + bv;
            if (relu) v = fmaxf(v, 0.f);
            if (out32) out32[(size_t)m * N_out + n] = v;
            if (out16) out16[(size_t)m * N_out + n] = (_Float16)v;
        }
    }
}

// ---------------------------------------------------------------------------
// Loss: log-softmax CE over 81 classes (mean over 512) +
//       smooth-L1 on positive ROIs (sum / 512). Single block.
// ---------------------------------------------------------------------------
__global__ void mrcnn_loss_kernel(const float* __restrict__ logits,  // 512 x 128
                                  const float* __restrict__ locp,    // 512 x 384
                                  const int*   __restrict__ label,
                                  const float* __restrict__ loc,     // 128 x 4
                                  float* __restrict__ out)
{
    __shared__ float scls[256];
    __shared__ float sbox[256];
    int t = threadIdx.x;

    float clsacc = 0.f;
    for (int n = t; n < N_ROI; n += 256) {
        const float* lr = logits + (size_t)n * NCLS_PAD;
        float mx = -1e30f;
        for (int k = 0; k < NUM_CLS; ++k) mx = fmaxf(mx, lr[k]);
        float se = 0.f;
        for (int k = 0; k < NUM_CLS; ++k) se += expf(lr[k] - mx);
        float lse = mx + logf(se);
        clsacc += lse - lr[label[n]];
    }

    float boxacc = 0.f;
    if (t < N_POS) {
        int ln = label[t];
        #pragma unroll
        for (int d = 0; d < 4; ++d) {
            float lt = locp[(size_t)t * NLOC_PAD + ln * 4 + d];
            float df = fabsf(lt - loc[t * 4 + d]);
            boxacc += (df < 1.f) ? 0.5f * df * df : df - 0.5f;
        }
    }

    scls[t] = clsacc;
    sbox[t] = boxacc;
    __syncthreads();
    for (int s2 = 128; s2 > 0; s2 >>= 1) {
        if (t < s2) { scls[t] += scls[t + s2]; sbox[t] += sbox[t + s2]; }
        __syncthreads();
    }
    if (t == 0)
        out[0] = scls[0] / (float)N_ROI + sbox[0] / (float)N_ROI;
}

// ---------------------------------------------------------------------------
extern "C" void kernel_launch(void* const* d_in, const int* in_sizes, int n_in,
                              void* d_out, int out_size, void* d_ws, size_t ws_size,
                              hipStream_t stream)
{
    const float* P2   = (const float*)d_in[0];
    const float* P3   = (const float*)d_in[1];
    const float* P4   = (const float*)d_in[2];
    const float* P5   = (const float*)d_in[3];
    const float* rois = (const float*)d_in[4];
    const int*   label= (const int*  )d_in[5];
    const float* loc  = (const float*)d_in[6];
    const float* W1   = (const float*)d_in[7];
    const float* b1   = (const float*)d_in[8];
    const float* W2   = (const float*)d_in[9];
    const float* b2   = (const float*)d_in[10];
    const float* Wcls = (const float*)d_in[11];
    const float* bcls = (const float*)d_in[12];
    const float* Wloc = (const float*)d_in[13];
    const float* bloc = (const float*)d_in[14];
    float* outp = (float*)d_out;

    // workspace carve-up (all region sizes are multiples of 256 bytes)
    char* ws = (char*)d_ws;
    size_t off = 0;
    _Float16* W1p  = (_Float16*)(ws + off); off += (size_t)K1  * HID      * 2; // 25.7 MB
    _Float16* W2p  = (_Float16*)(ws + off); off += (size_t)HID * HID      * 2; //  2.1 MB
    _Float16* Wcp  = (_Float16*)(ws + off); off += (size_t)HID * NCLS_PAD * 2;
    _Float16* Wlp  = (_Float16*)(ws + off); off += (size_t)HID * NLOC_PAD * 2;
    _Float16* x16  = (_Float16*)(ws + off); off += (size_t)N_ROI * K1     * 2; // 12.8 MB
    _Float16* h1   = (_Float16*)(ws + off); off += (size_t)N_ROI * HID    * 2;
    _Float16* h2   = (_Float16*)(ws + off); off += (size_t)N_ROI * HID    * 2;
    float*    logi = (float*)   (ws + off); off += (size_t)N_ROI * NCLS_PAD * 4;
    float*    locp = (float*)   (ws + off); off += (size_t)N_ROI * NLOC_PAD * 4;
    (void)ws_size; (void)in_sizes; (void)n_in; (void)out_size;

    // 1) pack weights into WMMA fragment layout (f32 -> f16)
    mrcnn_pack_w_kernel<<<2048, 256, 0, stream>>>(W1,   W1p, KT1, HID / 16, HID);
    mrcnn_pack_w_kernel<<<512,  256, 0, stream>>>(W2,   W2p, KT2, HID / 16, HID);
    mrcnn_pack_w_kernel<<<128,  256, 0, stream>>>(Wcls, Wcp, KT2, NCLS_PAD / 16, NUM_CLS);
    mrcnn_pack_w_kernel<<<384,  256, 0, stream>>>(Wloc, Wlp, KT2, NLOC_PAD / 16, NUM_CLS * 4);

    // 2) ROI-Align -> x (f16, 512 x 12544)
    mrcnn_roi_align_kernel<<<N_ROI, FEATC, 0, stream>>>(P2, P3, P4, P5, rois, x16);

    // 3) GEMM chain via v_wmma_f32_16x16x32_f16; wave = 16x64 tile (4 acc)
    {   // h1 = relu(x @ W1 + b1) : Mt=32, Ntg=16, Kt=392
        int Mt = N_ROI / 16, Ntg = (HID / 16) / 4, waves = Mt * Ntg;
        mrcnn_wmma_gemm_kernel<<<(waves + 3) / 4, 128, 0, stream>>>(
            x16, W1p, b1, nullptr, h1, Mt, Ntg, KT1, K1, HID, HID, 1);
    }
    {   // h2 = relu(h1 @ W2 + b2)
        int Mt = N_ROI / 16, Ntg = (HID / 16) / 4, waves = Mt * Ntg;
        mrcnn_wmma_gemm_kernel<<<(waves + 3) / 4, 128, 0, stream>>>(
            h1, W2p, b2, nullptr, h2, Mt, Ntg, KT2, HID, HID, HID, 1);
    }
    {   // logits = h2 @ Wcls + bcls (f32 out, padded to 128 cols)
        int Mt = N_ROI / 16, Ntg = (NCLS_PAD / 16) / 4, waves = Mt * Ntg;
        mrcnn_wmma_gemm_kernel<<<(waves + 3) / 4, 128, 0, stream>>>(
            h2, Wcp, bcls, logi, nullptr, Mt, Ntg, KT2, HID, NCLS_PAD, NUM_CLS, 0);
    }
    {   // locp = h2 @ Wloc + bloc (f32 out, padded to 384 cols)
        int Mt = N_ROI / 16, Ntg = (NLOC_PAD / 16) / 4, waves = Mt * Ntg;
        mrcnn_wmma_gemm_kernel<<<(waves + 3) / 4, 128, 0, stream>>>(
            h2, Wlp, bloc, locp, nullptr, Mt, Ntg, KT2, HID, NLOC_PAD, NUM_CLS * 4, 0);
    }

    // 4) losses -> scalar
    mrcnn_loss_kernel<<<1, 256, 0, stream>>>(logi, locp, label, loc, outp);
}
